// MoRE_73452530696712
// MI455X (gfx1250) — compile-verified, matrix operand order checked by reading, not run on hardware
//
#include <hip/hip_runtime.h>

typedef __attribute__((ext_vector_type(16))) __bf16 bf16x16;
typedef __attribute__((ext_vector_type(8)))  float  f32x8;
typedef __attribute__((ext_vector_type(4)))  float  f32x4;
typedef unsigned short u16;
typedef __attribute__((ext_vector_type(8)))  u16    u16x8;
typedef __attribute__((ext_vector_type(4)))  u16    u16x4;
typedef __attribute__((ext_vector_type(4)))  unsigned int u32x4;
typedef __attribute__((ext_vector_type(8)))  int    i32x8;
typedef __attribute__((ext_vector_type(4)))  int    i32x4;

#define B_TOT 32768
#define NE 16
#define NM 64
#define ND 256
#define ROWS_PER_WG 128

// LDS row strides in halfs (padding produced by TDM pad_enable)
#define KSTR 264   // 256 + 8
#define VSTR 72    // 64 + 8
#define ASTR 72    // 64 + 8

__device__ __forceinline__ u16 f2bf(float f) {
  union { float f; unsigned u; } c; c.f = f;
  unsigned u = c.u;
  unsigned r = u + 0x7FFFu + ((u >> 16) & 1u);   // round-to-nearest-even
  return (u16)(r >> 16);
}

union FragU { u16 h[16]; u16x8 q[2]; bf16x16 v; };

// A/B fragment loader for 16-bit 16x32 operands from row-major LDS.
__device__ __forceinline__ bf16x16 lds_frag(const u16* base, int row, int stride,
                                            int kbase, int lane) {
  int r  = row + (lane & 15);
  int k0 = kbase + ((lane >> 4) << 3);
  const u16* p = base + r * stride + k0;
  FragU u;
  u.q[0] = *(const u16x8*)(p);
  u.q[1] = *(const u16x8*)(p + 16);
  return u.v;
}

// x A-fragment straight from global fp32 (converted to bf16).
__device__ __forceinline__ bf16x16 x_frag(const float* __restrict__ x, int growbase,
                                          int kstep, int lane) {
  int r  = growbase + (lane & 15);
  int k0 = kstep * 32 + ((lane >> 4) << 3);
  const float* p = x + (size_t)r * ND + k0;
  f32x4 a  = *(const f32x4*)(p);
  f32x4 b  = *(const f32x4*)(p + 4);
  f32x4 c2 = *(const f32x4*)(p + 16);
  f32x4 d  = *(const f32x4*)(p + 20);
  FragU u;
#pragma unroll
  for (int i = 0; i < 4; i++) {
    u.h[i]      = f2bf(a[i]);
    u.h[4 + i]  = f2bf(b[i]);
    u.h[8 + i]  = f2bf(c2[i]);
    u.h[12 + i] = f2bf(d[i]);
  }
  return u.v;
}

__device__ __forceinline__ unsigned lds_off(const void* p) {
  // Flat LDS address: aperture lives in addr[63:32]; addr[31:0] is the
  // workgroup-relative LDS byte offset (ISA 10.2 per-aperture calc).
  return (unsigned)(uintptr_t)p;
}

// Issue a 2-D TDM load: global (bf16, row-major, tensor_d0 x tensor_d1)
// -> LDS with pad_amount DWORD-units of padding every 2^pad_interval 8-byte units.
__device__ __forceinline__ void tdm_load_2d(unsigned lds_addr, const void* gaddr,
                                            unsigned tensor_d0, unsigned tensor_d1,
                                            unsigned stride0,
                                            unsigned pad_interval, unsigned pad_amount) {
  unsigned long long ga = (unsigned long long)(uintptr_t)gaddr;
  u32x4 g0;
  g0[0] = 1u;                                        // count=1, user descriptor
  g0[1] = lds_addr;                                  // lds_addr [63:32]
  g0[2] = (unsigned)(ga & 0xFFFFFFFFu);              // global_addr [95:64]
  g0[3] = (unsigned)((ga >> 32) & 0x01FFFFFFu) | (2u << 30);  // addr[56:32] | type=2
  i32x8 g1;
  g1[0] = (int)((1u << 16)                           // data_size = 2 bytes
              | (1u << 20)                           // pad_enable
              | (pad_interval << 22)
              | (pad_amount << 25));
  g1[1] = (int)((tensor_d0 & 0xFFFFu) << 16);        // tensor_dim0[15:0] @ bits 63:48
  g1[2] = (int)(((tensor_d0 >> 16) & 0xFFFFu) | ((tensor_d1 & 0xFFFFu) << 16));
  g1[3] = (int)(((tensor_d1 >> 16) & 0xFFFFu) | ((tensor_d0 & 0xFFFFu) << 16)); // tile_dim0 = tensor_d0
  g1[4] = (int)(tensor_d1 & 0xFFFFu);                // tile_dim1 = tensor_d1; tile_dim2 = 0
  g1[5] = (int)stride0;                              // tensor_dim0_stride[31:0]
  g1[6] = 0;
  g1[7] = 0;
  i32x4 gz = {0, 0, 0, 0};
#if __clang_major__ >= 23
  i32x8 gz8 = {0, 0, 0, 0, 0, 0, 0, 0};
  __builtin_amdgcn_tensor_load_to_lds(g0, g1, gz, gz, gz8, 0);
#else
  __builtin_amdgcn_tensor_load_to_lds(g0, g1, gz, gz, 0);
#endif
}

// ---- prep kernel: K -> bf16 row-major, V -> bf16 transposed [E,D,M] ----
__global__ __launch_bounds__(256) void convert_kv_kernel(
    const float* __restrict__ K, const float* __restrict__ V,
    u16* __restrict__ kbf, u16* __restrict__ vtb) {
  int idx = blockIdx.x * 256 + threadIdx.x;          // 65536 threads, 4 elems each
  size_t base = (size_t)idx * 4;
  f32x4 kf = *(const f32x4*)(K + base);
  u16x4 kb;
#pragma unroll
  for (int j = 0; j < 4; j++) kb[j] = f2bf(kf[j]);
  *(u16x4*)(kbf + base) = kb;
  f32x4 vf = *(const f32x4*)(V + base);
  int e   = (int)(base / (NM * ND));
  int rem = (int)(base % (NM * ND));
  int m = rem / ND;
  int d = rem % ND;
#pragma unroll
  for (int j = 0; j < 4; j++)
    vtb[(size_t)e * ND * NM + (size_t)(d + j) * NM + m] = f2bf(vf[j]);
}

__global__ __launch_bounds__(256) void more_expert_kernel(
    const float* __restrict__ x, const u16* __restrict__ kbf,
    const u16* __restrict__ vtb, const float* __restrict__ wg,
    const float* __restrict__ bg, float* __restrict__ out) {
  __shared__ u16 sK[2][NM * KSTR];            // K_e bf16, double buffered  (2x33.8 KB)
  __shared__ u16 sVt[2][ND * VSTR];           // V_e^T bf16, double buffered(2x36.9 KB)
  __shared__ u16 sAt[ROWS_PER_WG * ASTR];     // attn tile bf16             (18.4 KB)

  const int tid  = threadIdx.x;
  const int wave = tid >> 5;
  const int lane = tid & 31;
  const int half = lane >> 4;
  const int l15  = lane & 15;
  const int blockRow = blockIdx.x * ROWS_PER_WG;
  const int rb = wave * 16;
  const int growbase = blockRow + rb;

  float* out_w = out;                                   // winners [B]
  float* out_s = out + B_TOT;                           // max scores [B]
  float* out_y = out + 2 * (size_t)B_TOT;               // y [B,E,D]
  float* out_g = out_y + (size_t)B_TOT * NE * ND;       // g [B,E]
  float* out_a = out_g + (size_t)B_TOT * NE;            // attn [B,E,M]

  const f32x8 zero = {0.f, 0.f, 0.f, 0.f, 0.f, 0.f, 0.f, 0.f};

  // x A-fragments, held in VGPRs across the whole expert loop
  bf16x16 xa[8];
#pragma unroll
  for (int ks = 0; ks < 8; ks++) xa[ks] = x_frag(x, growbase, ks, lane);

  // per-row ||x||
  float xnorm[16];
  for (int rr = 0; rr < 16; rr++) {
    const float* xr = x + (size_t)(growbase + rr) * ND + lane * 8;
    float s = 0.f;
#pragma unroll
    for (int j = 0; j < 8; j++) { float v = xr[j]; s += v * v; }
#pragma unroll
    for (int m = 16; m >= 1; m >>= 1) s += __shfl_xor(s, m, 32);
    xnorm[rr] = sqrtf(s);
  }

  float bestF[8];
  int   bestE[8];
#pragma unroll
  for (int r = 0; r < 8; r++) { bestF[r] = -3.4e38f; bestE[r] = 0; }

  // Prologue: TDM-load expert 0 into buffer 0 (wave 0 only; TDM ignores EXEC,
  // issued once per wave).
  if (wave == 0) {
    tdm_load_2d(lds_off(&sK[0][0]),  kbf, ND, NM, ND, 6, 3);  // 512B rows -> +16B pad
    tdm_load_2d(lds_off(&sVt[0][0]), vtb, NM, ND, NM, 4, 3);  // 128B rows -> +16B pad
  }

  int cur = 0;
  for (int e = 0; e < NE; e++) {
    if (wave == 0) {
      if (e + 1 < NE) {
        // prefetch next expert into the other buffer (its last readers passed
        // the end-of-iteration barrier of e-1 already)
        tdm_load_2d(lds_off(&sK[cur ^ 1][0]),  kbf + (size_t)(e + 1) * NM * ND,
                    ND, NM, ND, 6, 3);
        tdm_load_2d(lds_off(&sVt[cur ^ 1][0]), vtb + (size_t)(e + 1) * NM * ND,
                    NM, ND, NM, 4, 3);
        // TDM ops complete in order: <=2 outstanding => this expert's loads done
        __builtin_amdgcn_s_wait_tensorcnt(2);
      } else {
        __builtin_amdgcn_s_wait_tensorcnt(0);
      }
    }
    __syncthreads();   // releases buf[cur] to all waves
    const u16* cK  = &sK[cur][0];
    const u16* cVt = &sVt[cur][0];

    // ---- GEMM1: logits[16 x 64] = x_tile (16x256) * K_e^T ----
    f32x8 acc[4];
#pragma unroll
    for (int mt = 0; mt < 4; mt++) acc[mt] = zero;
#pragma unroll
    for (int ks = 0; ks < 8; ks++) {
#pragma unroll
      for (int mt = 0; mt < 4; mt++) {
        bf16x16 bk = lds_frag(cK, mt * 16, KSTR, ks * 32, lane);
        acc[mt] = __builtin_amdgcn_wmma_f32_16x16x32_bf16(
            false, xa[ks], false, bk, (short)0, acc[mt], false, false);
      }
    }

    // ---- softmax over M=64 per row ----
    float rsum[8];
#pragma unroll
    for (int r = 0; r < 8; r++) {
      float m = acc[0][r];
#pragma unroll
      for (int mt = 1; mt < 4; mt++) m = fmaxf(m, acc[mt][r]);
      m *= 0.0625f;  // 1/sqrt(256)
#pragma unroll
      for (int msk = 8; msk >= 1; msk >>= 1) m = fmaxf(m, __shfl_xor(m, msk, 32));
      float s = 0.f;
#pragma unroll
      for (int mt = 0; mt < 4; mt++) {
        float ev = __expf(acc[mt][r] * 0.0625f - m);
        acc[mt][r] = ev;
        s += ev;
      }
#pragma unroll
      for (int msk = 8; msk >= 1; msk >>= 1) s += __shfl_xor(s, msk, 32);
      rsum[r] = s;
    }
#pragma unroll
    for (int mt = 0; mt < 4; mt++) {
#pragma unroll
      for (int r = 0; r < 8; r++) {
        int rr = r + half * 8;
        float a = acc[mt][r] / rsum[r];
        int brow = growbase + rr;
        out_a[((size_t)brow * NE + e) * NM + mt * 16 + l15] = a;
        sAt[(rb + rr) * ASTR + mt * 16 + l15] = f2bf(a);
      }
    }

    // ---- GEMM2: y[16 x 256] = attn (16x64) * V_e (64x256) ----
    float dotp[8], yn2[8];
#pragma unroll
    for (int r = 0; r < 8; r++) { dotp[r] = 0.f; yn2[r] = 0.f; }
    bf16x16 af0 = lds_frag(sAt, rb, ASTR, 0, lane);
    bf16x16 af1 = lds_frag(sAt, rb, ASTR, 32, lane);
#pragma unroll
    for (int dt = 0; dt < 16; dt++) {
      bf16x16 bv0 = lds_frag(cVt, dt * 16, VSTR, 0, lane);
      bf16x16 bv1 = lds_frag(cVt, dt * 16, VSTR, 32, lane);
      f32x8 yacc = zero;
      yacc = __builtin_amdgcn_wmma_f32_16x16x32_bf16(
          false, af0, false, bv0, (short)0, yacc, false, false);
      yacc = __builtin_amdgcn_wmma_f32_16x16x32_bf16(
          false, af1, false, bv1, (short)0, yacc, false, false);
      int d = dt * 16 + l15;
#pragma unroll
      for (int r = 0; r < 8; r++) {
        int rr = r + half * 8;
        int brow = growbase + rr;
        float yv = yacc[r];
        out_y[((size_t)brow * NE + e) * ND + d] = yv;
        float xv = x[(size_t)brow * ND + d];
        dotp[r] += xv * yv;
        yn2[r]  += yv * yv;
      }
    }
#pragma unroll
    for (int r = 0; r < 8; r++) {
#pragma unroll
      for (int msk = 8; msk >= 1; msk >>= 1) {
        dotp[r] += __shfl_xor(dotp[r], msk, 32);
        yn2[r]  += __shfl_xor(yn2[r],  msk, 32);
      }
    }

    // ---- gate, familiarity, running argmax ----
    const float* wge = wg + (size_t)e * ND;
    float wgv[8];
#pragma unroll
    for (int j = 0; j < 8; j++) wgv[j] = wge[lane * 8 + j];
    float bge = bg[e];
    for (int rr = 0; rr < 16; rr++) {
      const float* xr = x + (size_t)(growbase + rr) * ND + lane * 8;
      float s = 0.f;
#pragma unroll
      for (int j = 0; j < 8; j++) s += xr[j] * wgv[j];
#pragma unroll
      for (int m = 16; m >= 1; m >>= 1) s += __shfl_xor(s, m, 32);
      float g = 1.f / (1.f + __expf(-(s + bge)));
      if ((rr >> 3) == half) {
        if (l15 == 0) out_g[(size_t)(growbase + rr) * NE + e] = g;
        int r = rr & 7;
        float cosv = dotp[r] / (xnorm[rr] * sqrtf(yn2[r]) + 1e-8f);
        float f = g * cosv;
        if (f > bestF[r]) { bestF[r] = f; bestE[r] = e; }
      }
    }
    __syncthreads();   // all waves done reading buf[cur] before TDM may overwrite
    cur ^= 1;
  }

  // ---- winners and max scores ----
  if (l15 == 0) {
#pragma unroll
    for (int r = 0; r < 8; r++) {
      int brow = growbase + half * 8 + r;
      out_w[brow] = (float)bestE[r];
      out_s[brow] = bestF[r];
    }
  }
}

extern "C" void kernel_launch(void* const* d_in, const int* in_sizes, int n_in,
                              void* d_out, int out_size, void* d_ws, size_t ws_size,
                              hipStream_t stream) {
  (void)in_sizes; (void)n_in; (void)out_size; (void)ws_size;
  const float* x  = (const float*)d_in[0];
  const float* K  = (const float*)d_in[1];
  const float* V  = (const float*)d_in[2];
  const float* wg = (const float*)d_in[3];
  const float* bg = (const float*)d_in[4];
  float* out = (float*)d_out;

  u16* kbf = (u16*)d_ws;                                  // [E,64,256] bf16 (512 KB)
  u16* vtb = (u16*)d_ws + (size_t)NE * NM * ND;           // [E,256,64] bf16 (512 KB)

  convert_kv_kernel<<<dim3(NE * NM * ND / 4 / 256), dim3(256), 0, stream>>>(K, V, kbf, vtb);
  more_expert_kernel<<<dim3(B_TOT / ROWS_PER_WG), dim3(256), 0, stream>>>(
      x, kbf, vtb, wg, bg, out);
}
